// Net_67302137528743
// MI455X (gfx1250) — matrix operand order repaired and checked by reading, not compiled, not run on hardware
//
#include <hip/hip_runtime.h>
#include <hip/hip_bf16.h>

#define SEQ      512
#define BATCH    65536
#define HID      2
#define DIM      256
#define MTILE    128         // batch rows per MLP workgroup
#define THREADS  256         // 8 waves (wave32)
#define STRIDE   264         // LDS activation row stride in halves (16B aligned rows, bank padding)
#define LAYER_ELEMS (DIM*DIM) // 65536 packed halves per 256x256 layer

typedef _Float16 half8  __attribute__((ext_vector_type(8)));
typedef _Float16 half16 __attribute__((ext_vector_type(16)));
typedef float    f32x8  __attribute__((ext_vector_type(8)));

union AFrag { half16 v; half8 h[2]; };

// ---------------------------------------------------------------------------
// Phase 1: RNN scan. Two batch elements per thread -> float4 (b128) streaming
// of x[t][2p..2p+1][:]. 256 MB read, HBM-bound (~11us @ 23.3 TB/s).
// ---------------------------------------------------------------------------
__global__ __launch_bounds__(256) void rnn_kernel(
    const float* __restrict__ x,
    const float* __restrict__ w_ih, const float* __restrict__ b_ih,
    const float* __restrict__ w_hh, const float* __restrict__ b_hh,
    float* __restrict__ hout)
{
    int p = blockIdx.x * blockDim.x + threadIdx.x;   // 0 .. BATCH/2-1
    if (p >= BATCH / 2) return;

    const float w00 = w_ih[0], w01 = w_ih[1], w10 = w_ih[2], w11 = w_ih[3];
    const float u00 = w_hh[0], u01 = w_hh[1], u10 = w_hh[2], u11 = w_hh[3];
    const float c0 = b_ih[0] + b_hh[0];
    const float c1 = b_ih[1] + b_hh[1];

    const float4* xp = (const float4*)x;             // [t*(BATCH/2) + p]
    float a0 = 0.0f, a1 = 0.0f;                      // h of batch 2p
    float g0 = 0.0f, g1 = 0.0f;                      // h of batch 2p+1

    float4 xt = xp[p];
    for (int t = 0; t < SEQ - 1; ++t) {
        float4 xn = xp[(size_t)(t + 1) * (BATCH / 2) + p];   // prefetch next step
        float pa0 = fmaf(xt.x, w00, fmaf(xt.y, w01, fmaf(a0, u00, fmaf(a1, u01, c0))));
        float pa1 = fmaf(xt.x, w10, fmaf(xt.y, w11, fmaf(a0, u10, fmaf(a1, u11, c1))));
        float pg0 = fmaf(xt.z, w00, fmaf(xt.w, w01, fmaf(g0, u00, fmaf(g1, u01, c0))));
        float pg1 = fmaf(xt.z, w10, fmaf(xt.w, w11, fmaf(g0, u10, fmaf(g1, u11, c1))));
        a0 = tanhf(pa0); a1 = tanhf(pa1);
        g0 = tanhf(pg0); g1 = tanhf(pg1);
        xt = xn;
    }
    float pa0 = fmaf(xt.x, w00, fmaf(xt.y, w01, fmaf(a0, u00, fmaf(a1, u01, c0))));
    float pa1 = fmaf(xt.x, w10, fmaf(xt.y, w11, fmaf(a0, u10, fmaf(a1, u11, c1))));
    float pg0 = fmaf(xt.z, w00, fmaf(xt.w, w01, fmaf(g0, u00, fmaf(g1, u01, c0))));
    float pg1 = fmaf(xt.z, w10, fmaf(xt.w, w11, fmaf(g0, u10, fmaf(g1, u11, c1))));

    float4 r;
    r.x = tanhf(pa0); r.y = tanhf(pa1);
    r.z = tanhf(pg0); r.w = tanhf(pg1);
    ((float4*)hout)[p] = r;                          // h[2p][0..1], h[2p+1][0..1]
}

// ---------------------------------------------------------------------------
// Weight repack: w2/w3/w4 (f32 [256][256], out = act @ w^T) into f16 WMMA
// B-fragment order. B is 32x16 (KxN) per wmma_f32_16x16x32_f16:
//   lane L: N = L%16 ; halves j=0..15 hold K = ks*32 + 8*(L/16) + (j&7) + 16*(j/8)
// Packed index: (((nt*8 + ks)*32 + lane)*16 + j) within a layer.
// ---------------------------------------------------------------------------
__global__ __launch_bounds__(256) void prep_weights(
    const float* __restrict__ w2, const float* __restrict__ w3,
    const float* __restrict__ w4, _Float16* __restrict__ packed)
{
    int idx = blockIdx.x * blockDim.x + threadIdx.x;
    if (idx >= 3 * LAYER_ELEMS) return;
    int l = idx >> 16;          // layer 0..2
    int t = idx & (LAYER_ELEMS - 1);
    int j    =  t        & 15;
    int lane = (t >> 4)  & 31;
    int ks   = (t >> 9)  & 7;
    int nt   =  t >> 12;
    int n = nt * 16 + (lane & 15);
    int k = ks * 32 + ((lane >> 4) << 3) + (j & 7) + ((j >> 3) << 4);
    const float* w = (l == 0) ? w2 : (l == 1) ? w3 : w4;
    packed[idx] = (_Float16)w[n * DIM + k];   // B[k][n] = w[n][k]
}

// ---------------------------------------------------------------------------
// Async-stage one packed 256x256 layer (128 KB) from global into LDS, shared
// by all 8 waves. GLOBAL_LOAD_ASYNC_TO_LDS_B128: vdst = LDS byte offset
// (low 32 bits of the flat LDS-aperture address), vaddr = 64-bit global.
// Tracked by ASYNCcnt; completed with s_wait_asynccnt + workgroup barrier.
// ---------------------------------------------------------------------------
__device__ __forceinline__ void stage_weights_issue(
    const _Float16* __restrict__ wpack, _Float16* __restrict__ bstage, int tid)
{
    unsigned dbase = (unsigned)(uintptr_t)bstage;
#pragma unroll
    for (int it = 0; it < 32; ++it) {
        unsigned byteOff = (unsigned)(it * THREADS + tid) * 16u;   // 16B per lane per iter
        unsigned d = dbase + byteOff;
        const void* s = (const char*)wpack + byteOff;
        asm volatile("global_load_async_to_lds_b128 %0, %1, off"
                     :: "v"(d), "v"(s) : "memory");
    }
}

__device__ __forceinline__ void stage_weights_wait()
{
    asm volatile("s_wait_asynccnt 0x0" ::: "memory");
}

// ---------------------------------------------------------------------------
// One 256x256 f16 WMMA layer: src LDS act -> dst LDS act, B from LDS stage,
// bias + ReLU fused. Wave wv owns rows [wv*16, wv*16+16).
//   A (ISA layout): lane<16: K = base+{0..7, 16..23}; lane>=16: +8  -> 2x b128
//   B frag: 32B contiguous per lane in staged layout                -> 2x b128
//   C: VGPR r -> row r + 8*(lane>=16), col = lane%16.
// N-tile loop unrolled 4x with a single advancing base pointer so every DS
// offset fits the 16-bit immediate (<= ~32 KB per group): no address-register
// wall. Per tile: 16 independent ds_load_b128, then two independent 4-deep
// wmma chains (even/odd K) merged in the epilogue -> 2x XDL ILP.
// ---------------------------------------------------------------------------
__device__ __forceinline__ void mlp_layer(
    const _Float16* __restrict__ src, _Float16* __restrict__ dst,
    const _Float16* __restrict__ bstage, const float* __restrict__ bias,
    int lane, int wv)
{
    const int rowA = wv * 16 + (lane & 15);
    const int klo  = (lane < 16) ? 0 : 8;

    AFrag a[8];
    {
        const _Float16* ab = src + rowA * STRIDE + klo;
#pragma unroll
        for (int ks = 0; ks < 8; ++ks) {
            a[ks].h[0] = *(const half8*)(ab + ks * 32);
            a[ks].h[1] = *(const half8*)(ab + ks * 32 + 16);
        }
    }

    const int rowC = wv * 16 + ((lane >> 4) << 3);
    const int ncol = lane & 15;

    const _Float16* bp   = bstage + lane * 16;    // advancing fragment base
    _Float16*       db   = dst + rowC * STRIDE + ncol;
    const float*    bsp  = bias + ncol;

#pragma unroll 4
    for (int nt = 0; nt < 16; ++nt) {
        AFrag b[8];
#pragma unroll
        for (int ks = 0; ks < 8; ++ks) {          // 16 independent ds_load_b128
            b[ks].h[0] = *(const half8*)(bp + ks * 512);
            b[ks].h[1] = *(const half8*)(bp + ks * 512 + 8);
        }

        f32x8 acc0 = {};
        f32x8 acc1 = {};
#pragma unroll
        for (int ks = 0; ks < 8; ks += 2) {       // two independent chains
            acc0 = __builtin_amdgcn_wmma_f32_16x16x32_f16(
                false, a[ks + 0].v, false, b[ks + 0].v, (short)0, acc0, false, false);
            acc1 = __builtin_amdgcn_wmma_f32_16x16x32_f16(
                false, a[ks + 1].v, false, b[ks + 1].v, (short)0, acc1, false, false);
        }

        float bv = *bsp;
#pragma unroll
        for (int r = 0; r < 8; ++r) {
            float v = acc0[r] + acc1[r] + bv;
            db[r * STRIDE] = (_Float16)fmaxf(v, 0.0f);
        }

        bp  += 8 * 512;                           // next N-tile's fragments
        db  += 16;
        bsp += 16;
    }
}

__global__ __launch_bounds__(THREADS) void mlp_kernel(
    const float* __restrict__ hin,
    const _Float16* __restrict__ packed,
    const float* __restrict__ w1, const float* __restrict__ b1,
    const float* __restrict__ b2, const float* __restrict__ b3,
    const float* __restrict__ b4,
    const float* __restrict__ w5, const float* __restrict__ b5,
    float* __restrict__ out)
{
    extern __shared__ __align__(16) _Float16 smem[];
    _Float16* buf0   = smem;                          // 128 x 264 halves
    _Float16* buf1   = smem + MTILE * STRIDE;         // 128 x 264 halves
    _Float16* bstage = smem + 2 * MTILE * STRIDE;     // 65536 halves (128 KB)

    const int tid  = threadIdx.x;
    const int lane = tid & 31;
    const int wv   = tid >> 5;
    const int rowBase = blockIdx.x * MTILE;

    // Kick off async staging of layer-2 weights; overlap with scalar layer 1.
    stage_weights_issue(packed + (size_t)0 * LAYER_ELEMS, bstage, tid);

    // ---- Layer 1: 2 -> 256, scalar (too thin for WMMA) -> buf0 (f16) ----
    {
        int row = tid >> 1;                 // 0..127
        int n0  = (tid & 1) * (DIM / 2);    // half the columns each
        float h0 = hin[(rowBase + row) * 2 + 0];
        float h1 = hin[(rowBase + row) * 2 + 1];
        _Float16* dstp = buf0 + row * STRIDE;
        for (int n = n0; n < n0 + DIM / 2; ++n) {
            float v = fmaf(h0, w1[n * 2 + 0], fmaf(h1, w1[n * 2 + 1], b1[n]));
            dstp[n] = (_Float16)fmaxf(v, 0.0f);
        }
    }
    stage_weights_wait();
    __syncthreads();

    // ---- Layer 2 ----
    mlp_layer(buf0, buf1, bstage, b2, lane, wv);
    __syncthreads();

    // ---- Layer 3 ----
    stage_weights_issue(packed + (size_t)1 * LAYER_ELEMS, bstage, tid);
    stage_weights_wait();
    __syncthreads();
    mlp_layer(buf1, buf0, bstage, b3, lane, wv);
    __syncthreads();

    // ---- Layer 4 ----
    stage_weights_issue(packed + (size_t)2 * LAYER_ELEMS, bstage, tid);
    stage_weights_wait();
    __syncthreads();
    mlp_layer(buf0, buf1, bstage, b4, lane, wv);
    __syncthreads();

    // ---- Layer 5: 256 -> 2, scalar reduction per (row, j) ----
    {
        int row = tid >> 1;                 // 0..127
        int j   = tid & 1;
        const _Float16* act = buf1 + row * STRIDE;
        const float* wr = w5 + j * DIM;
        float acc = b5[j];
        for (int n = 0; n < DIM; ++n)
            acc = fmaf((float)act[n], wr[n], acc);
        out[(rowBase + row) * 2 + j] = acc;
    }
}

// ---------------------------------------------------------------------------
extern "C" void kernel_launch(void* const* d_in, const int* in_sizes, int n_in,
                              void* d_out, int out_size, void* d_ws, size_t ws_size,
                              hipStream_t stream) {
    const float* x    = (const float*)d_in[0];
    const float* w_ih = (const float*)d_in[1];
    const float* b_ih = (const float*)d_in[2];
    const float* w_hh = (const float*)d_in[3];
    const float* b_hh = (const float*)d_in[4];
    const float* w1   = (const float*)d_in[5];
    const float* b1   = (const float*)d_in[6];
    const float* w2   = (const float*)d_in[7];
    const float* b2   = (const float*)d_in[8];
    const float* w3   = (const float*)d_in[9];
    const float* b3   = (const float*)d_in[10];
    const float* w4   = (const float*)d_in[11];
    const float* b4   = (const float*)d_in[12];
    const float* w5   = (const float*)d_in[13];
    const float* b5   = (const float*)d_in[14];
    float* out = (float*)d_out;

    // workspace: [h f32 65536x2][packed f16 3x65536]
    float*    h      = (float*)d_ws;
    _Float16* packed = (_Float16*)((char*)d_ws + (size_t)BATCH * HID * sizeof(float));

    prep_weights<<<(3 * LAYER_ELEMS + 255) / 256, 256, 0, stream>>>(w2, w3, w4, packed);
    rnn_kernel<<<(BATCH / 2) / 256, 256, 0, stream>>>(x, w_ih, b_ih, w_hh, b_hh, h);

    // 2 act buffers (66 KB) + weight stage (128 KB) = 260 KB of 320 KB WGP LDS
    size_t lds_bytes = (size_t)(2 * MTILE * STRIDE + LAYER_ELEMS) * sizeof(_Float16);
    mlp_kernel<<<BATCH / MTILE, THREADS, lds_bytes, stream>>>(
        h, packed, w1, b1, b2, b3, b4, w5, b5, out);
}
